// CausalSelfAttention_57844619543075
// MI455X (gfx1250) — matrix operand order repaired and checked by reading, not compiled, not run on hardware
//
#include <hip/hip_runtime.h>
#include <hip/hip_bf16.h>

// ---------------- types & constants ----------------
typedef __attribute__((ext_vector_type(16))) __bf16   v16bf;
typedef __attribute__((ext_vector_type(8)))  float    v8f;
typedef __attribute__((ext_vector_type(8)))  unsigned v8u;
typedef __attribute__((ext_vector_type(4)))  unsigned v4u;
typedef __attribute__((ext_vector_type(8)))  int      v8i;
typedef __attribute__((ext_vector_type(4)))  int      v4i;

static constexpr int BATCH = 2, T = 2048, C = 2048;
static constexpr int H = 16, KVH = 4, HD = 128, NREP = H / KVH;
static constexpr int NQ   = H * HD;        // 2048
static constexpr int NK   = KVH * HD;      // 512
static constexpr int NQKV = NQ + 2 * NK;   // 3072
static constexpr int M    = BATCH * T;     // 4096
static constexpr float SCALE = 0.08838834764831845f; // 1/sqrt(128)

// ---------------- helpers ----------------
__device__ __forceinline__ unsigned short f2bf(float f) {
  unsigned u = __builtin_bit_cast(unsigned, f);
  u += 0x7FFFu + ((u >> 16) & 1u);          // round-to-nearest-even
  return (unsigned short)(u >> 16);
}
__device__ __forceinline__ float bf2f(unsigned short h) {
  return __builtin_bit_cast(float, (unsigned)h << 16);
}
__device__ __forceinline__ unsigned pack2(unsigned short lo, unsigned short hi) {
  return (unsigned)lo | ((unsigned)hi << 16);
}
__device__ __forceinline__ v16bf as_bf16x16(v8u u) {
  return __builtin_bit_cast(v16bf, u);
}
__device__ __forceinline__ v8f zero8() {
  v8f z = {0.f, 0.f, 0.f, 0.f, 0.f, 0.f, 0.f, 0.f};
  return z;
}
__device__ __forceinline__ v8f wmma_bf16(v16bf a, v16bf b, v8f c) {
  // D = A(16x32 bf16) * B(32x16 bf16) + C(16x16 f32)
  return __builtin_amdgcn_wmma_f32_16x16x32_bf16(false, a, false, b, (short)0, c,
                                                 false, false);
}
// split fp32 pair into packed bf16 hi and bf16 residual (lo)
__device__ __forceinline__ void split2(float a, float b, unsigned& hi, unsigned& lo) {
  unsigned short ha = f2bf(a), hb = f2bf(b);
  hi = pack2(ha, hb);
  lo = pack2(f2bf(a - bf2f(ha)), f2bf(b - bf2f(hb)));
}

// ---- CDNA5 async global->LDS DMA (ASYNCcnt-tracked, no VGPR staging) ----
// VDST VGPR carries the LDS byte offset: the low 32 bits of a generic pointer
// to a __shared__ object are exactly that offset (aperture base is in the
// high half), per the FLAT aperture rules in CDNA5 ISA.
__device__ __forceinline__ void async_b128(const void* g, void* l) {
  unsigned lds = (unsigned)(size_t)l;
  unsigned long long ga = (unsigned long long)(size_t)g;
  asm volatile("global_load_async_to_lds_b128 %0, %1, off"
               :: "v"(lds), "v"(ga)
               : "memory");
}
__device__ __forceinline__ void wait_async0() {
  asm volatile("s_wait_asynccnt 0x0" ::: "memory");
}

// ---- CDNA5 Tensor Data Mover: 2D tile global->LDS ----
// D# per CDNA5 ISA; data_size=2 (4-byte units).
//   width_u  : tile/tensor dim0 in dwords
//   rows     : tile/tensor dim1
//   stride_u : tensor dim0 stride in dwords
// Toolchain is amdgpu-toolchain (clang-23): 6-arg builtin form
//   (uint32x4 g0, int32x8 g1, int32x4, int32x4, int32x8, i32 cpol)
__device__ __forceinline__ void tdm_load_2d(unsigned lds_addr,
                                            unsigned long long gaddr,
                                            unsigned width_u, unsigned rows,
                                            unsigned stride_u) {
  v4u g0;
  g0[0] = 1u;                                        // count=1, user descriptor
  g0[1] = lds_addr;                                  // LDS byte address
  g0[2] = (unsigned)(gaddr & 0xffffffffull);         // global_addr[31:0]
  g0[3] = (unsigned)((gaddr >> 32) & 0x1ffffffull)   // global_addr[56:32]
          | (2u << 30);                              // type=2 ("image")
  v8i g1;
  g1[0] = (int)(2u << 16);                           // wg_mask=0, data_size=4B
  g1[1] = (int)((width_u & 0xffffu) << 16);          // tensor_dim0[15:0]
  g1[2] = (int)((width_u >> 16) | ((rows & 0xffffu) << 16));   // dim0 hi | dim1 lo
  g1[3] = (int)((rows >> 16) | ((width_u & 0xffffu) << 16));   // dim1 hi | tile_dim0
  g1[4] = (int)(rows & 0xffffu);                     // tile_dim1, tile_dim2=0
  g1[5] = (int)stride_u;                             // tensor_dim0_stride[31:0]
  g1[6] = 0;                                         // stride0 hi | stride1 lo
  g1[7] = 0;
  v4i z4 = {0, 0, 0, 0};                             // 2D: groups 2/3 unused
  v8i z8 = {0, 0, 0, 0, 0, 0, 0, 0};
  __builtin_amdgcn_tensor_load_to_lds(g0, g1, z4, z4, z8, 0);
}

// =====================================================================
// Kernel 1: QKV = x @ W^T   (bf16x3 split-precision WMMA GEMM)
//   x: (M, C) fp32, W: (NQKV, C) fp32, qkv: (M, NQKV) fp32
//   block tile 128x128, 8 waves -> each wave 32x64 (2x4 16x16 subtiles)
// =====================================================================
__global__ __launch_bounds__(256) void qkv_gemm_kernel(
    const float* __restrict__ x, const float* __restrict__ w,
    float* __restrict__ qkv) {
  __shared__ unsigned Ah[128][16];  // packed bf16 pairs along K (hi)
  __shared__ unsigned Al[128][16];  // (lo residual)
  __shared__ unsigned Bh[128][16];
  __shared__ unsigned Bl[128][16];

  const int tid  = threadIdx.x;
  const int lane = tid & 31, wid = tid >> 5;
  const int g    = lane >> 4;       // lane half (selects K-half of frags)
  const int ln   = lane & 15;       // M (A-frag) or N (B/C/D-frag) index
  const int m0   = blockIdx.y * 128;
  const int n0   = blockIdx.x * 128;
  const int wm   = (wid & 3) * 32;  // wave M offset in block tile
  const int wn   = (wid >> 2) * 64; // wave N offset in block tile

  v8f acc[2][4];
#pragma unroll
  for (int i = 0; i < 2; ++i)
#pragma unroll
    for (int j = 0; j < 4; ++j) acc[i][j] = zero8();

  const int lrow = tid >> 1;         // 0..127 : tile row this thread loads
  const int lk   = (tid & 1) * 16;   // 0 or 16 : K sub-offset

  for (int k0 = 0; k0 < C; k0 += 32) {
    __syncthreads();
    if (k0 + 32 < C) {
      __builtin_prefetch(&x[(size_t)(m0 + lrow) * C + k0 + 32 + lk], 0, 1);
      __builtin_prefetch(&w[(size_t)(n0 + lrow) * C + k0 + 32 + lk], 0, 1);
    }
    // ---- stage A (x) and B (w) tiles as bf16 hi/lo packed pairs ----
    {
      const float4* xs = (const float4*)&x[(size_t)(m0 + lrow) * C + k0 + lk];
      const float4* ws = (const float4*)&w[(size_t)(n0 + lrow) * C + k0 + lk];
#pragma unroll
      for (int j4 = 0; j4 < 4; ++j4) {
        float4 xa = xs[j4];
        unsigned h0, l0, h1, l1;
        split2(xa.x, xa.y, h0, l0);
        split2(xa.z, xa.w, h1, l1);
        int p = (lk >> 1) + j4 * 2;
        Ah[lrow][p] = h0; Ah[lrow][p + 1] = h1;
        Al[lrow][p] = l0; Al[lrow][p + 1] = l1;
        float4 wa = ws[j4];
        split2(wa.x, wa.y, h0, l0);
        split2(wa.z, wa.w, h1, l1);
        Bh[lrow][p] = h0; Bh[lrow][p + 1] = h1;
        Bl[lrow][p] = l0; Bl[lrow][p + 1] = l1;
      }
    }
    __syncthreads();

    // ---- build fragments (ISA 7.12.2 16-bit A / B layouts) ----
    v16bf ah[2], al[2];
#pragma unroll
    for (int i = 0; i < 2; ++i) {
      v8u uh, ul;
#pragma unroll
      for (int v = 0; v < 4; ++v) {
        uh[v]     = Ah[wm + i * 16 + ln][4 * g + v];
        uh[4 + v] = Ah[wm + i * 16 + ln][8 + 4 * g + v];
        ul[v]     = Al[wm + i * 16 + ln][4 * g + v];
        ul[4 + v] = Al[wm + i * 16 + ln][8 + 4 * g + v];
      }
      ah[i] = as_bf16x16(uh);
      al[i] = as_bf16x16(ul);
    }
    v16bf bh[4], bl[4];
#pragma unroll
    for (int j = 0; j < 4; ++j) {
      v8u uh, ul;
#pragma unroll
      for (int v = 0; v < 8; ++v) {
        uh[v] = Bh[wn + j * 16 + ln][(v & 3) + 4 * g + 8 * (v >> 2)];
        ul[v] = Bl[wn + j * 16 + ln][(v & 3) + 4 * g + 8 * (v >> 2)];
      }
      bh[j] = as_bf16x16(uh);
      bl[j] = as_bf16x16(ul);
    }
    // ---- bf16x3: hi*hi + hi*lo + lo*hi, fp32 accumulate ----
#pragma unroll
    for (int i = 0; i < 2; ++i)
#pragma unroll
      for (int j = 0; j < 4; ++j) {
        acc[i][j] = wmma_bf16(ah[i], bh[j], acc[i][j]);
        acc[i][j] = wmma_bf16(ah[i], bl[j], acc[i][j]);
        acc[i][j] = wmma_bf16(al[i], bh[j], acc[i][j]);
      }
  }

  // ---- epilogue: C/D layout -> global (lane ln = col, VGPR r -> row r+8g)
#pragma unroll
  for (int i = 0; i < 2; ++i)
#pragma unroll
    for (int j = 0; j < 4; ++j)
#pragma unroll
      for (int r = 0; r < 8; ++r) {
        int row = m0 + wm + i * 16 + r + 8 * g;
        int col = n0 + wn + j * 16 + ln;
        qkv[(size_t)row * NQKV + col] = acc[i][j][r];
      }
}

// =====================================================================
// Kernel 2: RoPE on Q,K + repack Q/K/V to bf16 head-major layouts
//   qb: (B,T,H,HD) bf16 (uint-paired), kb/vb: (B,T,KVH,HD) bf16
// =====================================================================
__global__ __launch_bounds__(256) void rope_split_kernel(
    const float* __restrict__ qkv, const float* __restrict__ fcos,
    const float* __restrict__ fsin, unsigned* __restrict__ qb,
    unsigned* __restrict__ kb, unsigned* __restrict__ vb) {
  int idx = blockIdx.x * 256 + threadIdx.x;  // pair index
  if (idx >= M * (NQKV / 2)) return;
  int row = idx / (NQKV / 2);
  int col = (idx - row * (NQKV / 2)) * 2;
  int t = row & (T - 1);
  float e = qkv[(size_t)row * NQKV + col];
  float o = qkv[(size_t)row * NQKV + col + 1];
  if (col < NQ) {                       // Q with RoPE
    int h = col >> 7, d = col & 127, i = d >> 1;
    float cc = fcos[t * 64 + i], ss = fsin[t * 64 + i];
    qb[((size_t)row * H + h) * 64 + (d >> 1)] =
        pack2(f2bf(e * cc - o * ss), f2bf(e * ss + o * cc));
  } else if (col < NQ + NK) {           // K with RoPE
    int c2 = col - NQ;
    int kh = c2 >> 7, d = c2 & 127, i = d >> 1;
    float cc = fcos[t * 64 + i], ss = fsin[t * 64 + i];
    kb[((size_t)row * KVH + kh) * 64 + (d >> 1)] =
        pack2(f2bf(e * cc - o * ss), f2bf(e * ss + o * cc));
  } else {                              // V passthrough
    int c2 = col - NQ - NK;
    int vh = c2 >> 7, d = c2 & 127;
    vb[((size_t)row * KVH + vh) * 64 + (d >> 1)] = pack2(f2bf(e), f2bf(o));
  }
}

// =====================================================================
// Kernel 3: causal GQA flash attention (TDM + async LDS staged)
//   grid (T/128, H, B); 8 waves, each owns 16 q-rows of a 128-row block
// =====================================================================
__global__ __launch_bounds__(256) void flash_attn_kernel(
    const unsigned* __restrict__ qb, const unsigned* __restrict__ kb,
    const unsigned* __restrict__ vb, float* __restrict__ out) {
  __shared__ unsigned Qs[128][64];          // [qrow][dpair]     32 KB
  __shared__ unsigned Ks[64][64];           // [key ][dpair]     16 KB
  __shared__ unsigned Vs[64][64];           // [key ][dpair]     16 KB
  __shared__ unsigned short Ps[8][16][64];  // per-wave P        16 KB

  const int tid  = threadIdx.x;
  const int lane = tid & 31, wid = tid >> 5;
  const int g    = lane >> 4, ln = lane & 15;
  const int qb0  = blockIdx.x * 128;
  const int h    = blockIdx.y;
  const int b    = blockIdx.z;
  const int kvh  = h / NREP;

  // ---- stage Q block: per-lane async DMA, 8x b128 per thread ----
  {
    const unsigned* qsrc = qb + ((size_t)(b * T + qb0) * H + h) * 64;
#pragma unroll
    for (int it = 0; it < 8; ++it) {
      int c = it * 256 + tid;        // 2048 chunks of 16B
      int r = c >> 4, q4 = (c & 15) * 4;
      async_b128(qsrc + (size_t)r * (H * 64) + q4, &Qs[r][q4]);
    }
  }
  wait_async0();
  __syncthreads();

  // ---- Q A-frags: 4 chunks of K(dim)=32 ----
  v16bf qf[4];
#pragma unroll
  for (int f = 0; f < 4; ++f) {
    v8u u;
#pragma unroll
    for (int v = 0; v < 4; ++v) {
      u[v]     = Qs[wid * 16 + ln][f * 16 + 4 * g + v];
      u[4 + v] = Qs[wid * 16 + ln][f * 16 + 8 + 4 * g + v];
    }
    qf[f] = as_bf16x16(u);
  }

  v8f o_acc[8];
#pragma unroll
  for (int d = 0; d < 8; ++d) o_acc[d] = zero8();
  float mrow[8], lrow[8];
#pragma unroll
  for (int r = 0; r < 8; ++r) { mrow[r] = -3.0e38f; lrow[r] = 0.f; }

  const int nkb = (qb0 / 64) + 2;  // key blocks covering [0, qb0+128)
  for (int k64 = 0; k64 < nkb; ++k64) {
    __syncthreads();  // all waves done reading previous K/V tiles
    // ---- stage K + V tiles with the Tensor Data Mover (wave 0 only;
    //      TDM ignores EXEC, one issuing wave = one DMA per tile) ----
    const unsigned* ksrc = kb + ((size_t)(b * T + k64 * 64) * KVH + kvh) * 64;
    const unsigned* vsrc = vb + ((size_t)(b * T + k64 * 64) * KVH + kvh) * 64;
    if (wid == 0) {
      // 64 rows x 64 dwords tile, row stride KVH*64 dwords
      tdm_load_2d((unsigned)(size_t)&Ks[0][0],
                  (unsigned long long)(size_t)ksrc, 64u, 64u, (unsigned)(KVH * 64));
      tdm_load_2d((unsigned)(size_t)&Vs[0][0],
                  (unsigned long long)(size_t)vsrc, 64u, 64u, (unsigned)(KVH * 64));
      __builtin_amdgcn_s_wait_tensorcnt(0);
    }
    __syncthreads();  // tiles visible to all waves

    // ---- S = Q K^T (4 key-subtiles x 4 d-chunks of WMMA) ----
    v8f sreg[4];
#pragma unroll
    for (int j = 0; j < 4; ++j) {
      v8f s = zero8();
#pragma unroll
      for (int f = 0; f < 4; ++f) {
        v8u u;
#pragma unroll
        for (int v = 0; v < 8; ++v)
          u[v] = Ks[j * 16 + ln][f * 16 + (v & 3) + 4 * g + 8 * (v >> 2)];
        s = wmma_bf16(qf[f], as_bf16x16(u), s);
      }
      sreg[j] = s;
    }

    // ---- scale + causal mask ----
#pragma unroll
    for (int j = 0; j < 4; ++j) {
      int colj = k64 * 64 + j * 16 + ln;
#pragma unroll
      for (int r = 0; r < 8; ++r) {
        int trow = qb0 + wid * 16 + r + 8 * g;
        float v = sreg[j][r] * SCALE;
        sreg[j][r] = (colj <= trow) ? v : -3.0e38f;
      }
    }

    // ---- online softmax (16-lane butterflies match C-layout halves) ----
#pragma unroll
    for (int r = 0; r < 8; ++r) {
      float mx = fmaxf(fmaxf(sreg[0][r], sreg[1][r]),
                       fmaxf(sreg[2][r], sreg[3][r]));
#pragma unroll
      for (int off = 8; off >= 1; off >>= 1)
        mx = fmaxf(mx, __shfl_xor(mx, off, 32));
      float mnew = fmaxf(mrow[r], mx);
      float corr = __expf(mrow[r] - mnew);
      float rs = 0.f;
#pragma unroll
      for (int j = 0; j < 4; ++j) {
        float p = __expf(sreg[j][r] - mnew);
        sreg[j][r] = p;
        rs += p;
      }
#pragma unroll
      for (int off = 8; off >= 1; off >>= 1) rs += __shfl_xor(rs, off, 32);
      lrow[r] = lrow[r] * corr + rs;
      mrow[r] = mnew;
#pragma unroll
      for (int d = 0; d < 8; ++d) o_acc[d][r] *= corr;
    }

    // ---- P: C-layout -> A-frag layout via per-wave LDS region ----
#pragma unroll
    for (int j = 0; j < 4; ++j)
#pragma unroll
      for (int r = 0; r < 8; ++r)
        Ps[wid][r + 8 * g][j * 16 + ln] = f2bf(sreg[j][r]);
    // same-wave DS ops are in-order; compiler inserts s_wait_dscnt

    v16bf pf[2];
    const unsigned* pu32 = (const unsigned*)&Ps[wid][0][0];  // [16][32] uints
#pragma unroll
    for (int f = 0; f < 2; ++f) {
      v8u u;
#pragma unroll
      for (int v = 0; v < 4; ++v) {
        u[v]     = pu32[ln * 32 + f * 16 + 4 * g + v];
        u[4 + v] = pu32[ln * 32 + f * 16 + 8 + 4 * g + v];
      }
      pf[f] = as_bf16x16(u);
    }

    // ---- O += P V : B-frags gathered as (key,key+1) 16-bit pairs ----
    const unsigned short* v16p = (const unsigned short*)&Vs[0][0];  // [64][128]
#pragma unroll
    for (int d = 0; d < 8; ++d)
#pragma unroll
      for (int f = 0; f < 2; ++f) {
        v8u u;
#pragma unroll
        for (int v = 0; v < 8; ++v) {
          int key = f * 32 + 2 * (v & 3) + 8 * g + 16 * (v >> 2);
          int dd  = d * 16 + ln;
          u[v] = pack2(v16p[key * 128 + dd], v16p[(key + 1) * 128 + dd]);
        }
        o_acc[d] = wmma_bf16(pf[f], as_bf16x16(u), o_acc[d]);
      }
  }

  // ---- normalize + store ----
#pragma unroll
  for (int d = 0; d < 8; ++d)
#pragma unroll
    for (int r = 0; r < 8; ++r) {
      int trow = qb0 + wid * 16 + r + 8 * g;
      int col  = h * HD + d * 16 + ln;
      out[(size_t)(b * T + trow) * C + col] = o_acc[d][r] / lrow[r];
    }
}

// =====================================================================
// launch
// =====================================================================
extern "C" void kernel_launch(void* const* d_in, const int* in_sizes, int n_in,
                              void* d_out, int out_size, void* d_ws,
                              size_t ws_size, hipStream_t stream) {
  const float* x    = (const float*)d_in[0];
  const float* wqkv = (const float*)d_in[1];
  const float* fcos = (const float*)d_in[2];
  const float* fsin = (const float*)d_in[3];
  float* out = (float*)d_out;

  // workspace layout
  float* qkv   = (float*)d_ws;                                   // M*NQKV f32
  unsigned* qb = (unsigned*)(qkv + (size_t)M * NQKV);            // M*NQ/2
  unsigned* kb = qb + (size_t)M * (NQ / 2);                      // M*NK/2
  unsigned* vb = kb + (size_t)M * (NK / 2);                      // M*NK/2

  // 1) QKV projection (bf16x3 WMMA GEMM)
  dim3 g1(NQKV / 128, M / 128, 1);
  qkv_gemm_kernel<<<g1, 256, 0, stream>>>(x, wqkv, qkv);

  // 2) RoPE + bf16 repack
  int npairs = M * (NQKV / 2);
  rope_split_kernel<<<(npairs + 255) / 256, 256, 0, stream>>>(qkv, fcos, fsin,
                                                              qb, kb, vb);

  // 3) causal GQA flash attention (TDM + async LDS staged)
  dim3 g3(T / 128, H, BATCH);
  flash_attn_kernel<<<g3, 256, 0, stream>>>(qb, kb, vb, out);
}
